// Seq2Seq_62843961475565
// MI455X (gfx1250) — matrix operand order repaired and checked by reading, not compiled, not run on hardware
//
#include <hip/hip_runtime.h>
#include <hip/hip_bf16.h>
#include <cstddef>

// ---------------------------------------------------------------------------
// Seq2Seq (LSTM enc/dec + Bahdanau attention) for MI455X / gfx1250 (wave32).
//  * All matrix work: v_wmma_f32_16x16x32_bf16, fp32 accumulate.
//  * Weights transposed+converted to bf16 K-major once (contiguous 32B B loads).
//  * Activations feeding GEMMs are produced directly in bf16 -> no cvt in the
//    GEMM hot loop; A operand = two 16B loads per k-step.
//  * Each wave computes a 16x32 C tile: one A fetch feeds two WMMAs.
//  * Input projections (x @ Wih^T) batched over all timesteps into one big
//    GEMM (M=2048 / M=2016); recurrence loop only carries the h-GEMM.
//  * Bahdanau split: encproj = enc_out @ Wa_e^T + b computed once.
// ---------------------------------------------------------------------------

#define Bb   32
#define Ss   64
#define Tt   64
#define Ee   512
#define Hh   1024
#define VTt  32000
#define G4H  4096   // 4*H

typedef __bf16 bf16_t;
typedef __attribute__((ext_vector_type(16))) __bf16 v16bf;
typedef __attribute__((ext_vector_type(8)))  __bf16 v8bf;
typedef __attribute__((ext_vector_type(8)))  float  v8f;

// ---------------- transpose + convert: src (N,K) f32 -> dst (K,N) bf16 -----
__global__ __launch_bounds__(256) void transpose_cvt_bf16(
    const float* __restrict__ src, bf16_t* __restrict__ dst, int N, int K)
{
    __shared__ float tile[32][33];
    const int kb = blockIdx.x * 32;
    const int nb = blockIdx.y * 32;
    const int tx = threadIdx.x;          // 0..31
    const int ty = threadIdx.y;          // 0..7
    for (int i = ty; i < 32; i += 8) {
        int n = nb + i, k = kb + tx;
        tile[i][tx] = (n < N && k < K) ? src[(size_t)n * K + k] : 0.f;
    }
    __syncthreads();
    for (int i = ty; i < 32; i += 8) {
        int k = kb + i, n = nb + tx;
        if (k < K && n < N) dst[(size_t)k * N + n] = (bf16_t)tile[tx][i];
    }
}

// ---------------- embedding gather, emits bf16 (E = 512) -------------------
__global__ __launch_bounds__(256) void gather_emb_bf16(
    const int* __restrict__ tok, int Lsel, int tokStride,
    const float* __restrict__ table, bf16_t* __restrict__ out, int total)
{
    int idx = blockIdx.x * 256 + threadIdx.x;
    if (idx >= total) return;
    int e = idx & (Ee - 1);
    int r = idx >> 9;                    // / E
    int b = r / Lsel;
    int t = r - b * Lsel;
    int tk = tok[b * tokStride + t];
    out[idx] = (bf16_t)table[(size_t)tk * Ee + e];
}

// ---------------- zero helpers --------------------------------------------
__global__ __launch_bounds__(256) void zero_f32(float* p, int n)
{
    int i = blockIdx.x * 256 + threadIdx.x;
    if (i < n) p[i] = 0.f;
}
__global__ __launch_bounds__(256) void zero_strided(
    float* p, int rows, int cols, long long ld)
{
    int i = blockIdx.x * 256 + threadIdx.x;
    if (i >= rows * cols) return;
    int r = i / cols, cq = i - r * cols;
    p[(size_t)r * (size_t)ld + cq] = 0.f;
}

// ---------------- WMMA GEMM: C(M,N) (+)= A(M,K) @ Bt(K,N) + bias -----------
// A bf16 row-major (lda), Bt bf16 K-major (ldb), C fp32 (ldc).
// Block = 256 threads = 8 waves; wave w -> 16x32 C tile (two 16x16 WMMA
// accumulators sharing one A fetch). Block covers 16(M) x 256(N).
// All dims are exact multiples (M%16==0, N%256==0, K%32==0) -> no edge
// masking, EXEC stays all-ones around the WMMAs.
__global__ __launch_bounds__(256) void gemm_bf16_wmma(
    const bf16_t* __restrict__ A, int lda,
    const bf16_t* __restrict__ Bt, int ldb,
    const float* __restrict__ bias,
    float* __restrict__ C, int ldc,
    int K, int accumulate)
{
    const int wave = threadIdx.x >> 5;
    const int lane = threadIdx.x & 31;
    const int l15  = lane & 15;
    const bool hi  = lane >= 16;
    const int n0   = blockIdx.x * 256 + wave * 32;
    const int m0   = blockIdx.y * 16;

    // A layout (16-bit A-matrix 16x32): lo lanes hold K={0..7,16..23},
    // hi lanes hold K={8..15,24..31} of row M = m0 + (lane&15).
    const bf16_t* aptr = A + (size_t)(m0 + l15) * lda + (hi ? 8 : 0);
    // B layout (16-bit B 32x16): lane holds column N = n0 + (lane&15),
    // K={0..7,16..23} (lo) / K={8..15,24..31} (hi), matching A.
    const bf16_t* bptr = Bt + (size_t)(hi ? 8 : 0) * ldb + n0 + l15;

    v8f acc0 = {0.f, 0.f, 0.f, 0.f, 0.f, 0.f, 0.f, 0.f};
    v8f acc1 = {0.f, 0.f, 0.f, 0.f, 0.f, 0.f, 0.f, 0.f};

    for (int k0 = 0; k0 < K; k0 += 32) {
        v8bf alo = *(const v8bf*)(aptr + k0);
        v8bf ahi = *(const v8bf*)(aptr + k0 + 16);
        v16bf a  = __builtin_shufflevector(alo, ahi,
                       0, 1, 2, 3, 4, 5, 6, 7, 8, 9, 10, 11, 12, 13, 14, 15);
        const bf16_t* bk = bptr + (size_t)k0 * ldb;
        v16bf b0, b1;
#pragma unroll
        for (int s = 0; s < 16; ++s) {
            const size_t kk = (size_t)(s < 8 ? s : s + 8) * ldb;
            b0[s] = bk[kk];
            b1[s] = bk[kk + 16];
        }
        acc0 = __builtin_amdgcn_wmma_f32_16x16x32_bf16(
                   false, a, false, b0, (short)0, acc0, false, false);
        acc1 = __builtin_amdgcn_wmma_f32_16x16x32_bf16(
                   false, a, false, b1, (short)0, acc1, false, false);
    }

    // C/D layout: lane n = n0+(lane&15); VGPR r -> M = m0 + (hi?8:0) + r.
    const int   cn0 = n0 + l15;
    const int   cm0 = m0 + (hi ? 8 : 0);
    const float bv0 = bias ? bias[cn0]      : 0.f;
    const float bv1 = bias ? bias[cn0 + 16] : 0.f;
    float* cp = C + (size_t)cm0 * ldc + cn0;
#pragma unroll
    for (int r = 0; r < 8; ++r) {
        float v0 = acc0[r] + bv0;
        float v1 = acc1[r] + bv1;
        if (accumulate) {
            cp[(size_t)r * ldc]      += v0;
            cp[(size_t)r * ldc + 16] += v1;
        } else {
            cp[(size_t)r * ldc]      = v0;
            cp[(size_t)r * ldc + 16] = v1;
        }
    }
}

// ---------------- LSTM cell (gate order i,f,g,o) ---------------------------
// gates = gatesH[b, 4H] + gatesX[b*gxStride + 4H slice]; biases already folded
// into the GEMMs. Emits h in bf16 (GEMM A operand); optional enc_out slices.
__global__ __launch_bounds__(256) void lstm_cell(
    const float* __restrict__ gatesH,
    const float* __restrict__ gatesX, long long gxStride,
    float* __restrict__ c, bf16_t* __restrict__ hbf,
    float* __restrict__ encout, bf16_t* __restrict__ encoutbf, int ldslice)
{
    int idx = blockIdx.x * 256 + threadIdx.x;
    if (idx >= Bb * Hh) return;
    int b = idx >> 10, j = idx & (Hh - 1);
    const float* gh = gatesH + (size_t)b * G4H;
    const float* gx = gatesX + (size_t)b * gxStride;
    float ig = gh[j]          + gx[j];
    float fg = gh[Hh + j]     + gx[Hh + j];
    float gg = gh[2 * Hh + j] + gx[2 * Hh + j];
    float og = gh[3 * Hh + j] + gx[3 * Hh + j];
    float si = 1.f / (1.f + expf(-ig));
    float sf = 1.f / (1.f + expf(-fg));
    float so = 1.f / (1.f + expf(-og));
    float cn = sf * c[idx] + si * tanhf(gg);
    c[idx] = cn;
    float hn = so * tanhf(cn);
    hbf[idx] = (bf16_t)hn;
    if (encout) {
        encout[(size_t)b * ldslice + j]   = hn;
        encoutbf[(size_t)b * ldslice + j] = (bf16_t)hn;
    }
}

// ---------------- attention scores: scores[r] = sum_h tanh(hp+ep)*v --------
__global__ __launch_bounds__(256) void attn_scores(
    const float* __restrict__ hproj, const float* __restrict__ encproj,
    const float* __restrict__ vW, float* __restrict__ scores)
{
    const int r = blockIdx.x;            // 0 .. B*S-1
    const int b = r >> 6;                // / S
    const int tid = threadIdx.x;
    float p = 0.f;
    for (int hh = tid; hh < Hh; hh += 256)
        p += tanhf(hproj[(size_t)b * Hh + hh] + encproj[(size_t)r * Hh + hh]) * vW[hh];
    __shared__ float red[256];
    red[tid] = p;
    __syncthreads();
    for (int s2 = 128; s2 > 0; s2 >>= 1) {
        if (tid < s2) red[tid] += red[tid + s2];
        __syncthreads();
    }
    if (tid == 0) scores[r] = red[0];
}

// ---------------- softmax over S + context = attn @ enc_out (bf16 out) -----
__global__ __launch_bounds__(256) void softmax_context(
    const float* __restrict__ scores, const float* __restrict__ enc_out,
    bf16_t* __restrict__ context)
{
    const int b = blockIdx.x;
    const int tid = threadIdx.x;
    __shared__ float sc[Ss];
    __shared__ float w[Ss];
    if (tid < Ss) sc[tid] = scores[b * Ss + tid];
    __syncthreads();
    float mx = -3.4e38f;
    for (int s = 0; s < Ss; ++s) mx = fmaxf(mx, sc[s]);
    float sum = 0.f;
    for (int s = 0; s < Ss; ++s) sum += expf(sc[s] - mx);
    if (tid < Ss) w[tid] = expf(sc[tid] - mx) / sum;
    __syncthreads();
    for (int hh = tid; hh < Hh; hh += 256) {
        float acc = 0.f;
        const float* eo = enc_out + (size_t)b * Ss * Hh + hh;
        for (int s = 0; s < Ss; ++s) acc += w[s] * eo[(size_t)s * Hh];
        context[(size_t)b * Hh + hh] = (bf16_t)acc;
    }
}

// ===========================================================================
extern "C" void kernel_launch(void* const* d_in, const int* in_sizes, int n_in,
                              void* d_out, int out_size, void* d_ws, size_t ws_size,
                              hipStream_t stream)
{
    (void)in_sizes; (void)n_in; (void)out_size; (void)ws_size;

    const int*   src      = (const int*)  d_in[0];
    const int*   tgt      = (const int*)  d_in[1];
    const float* enc_emb  = (const float*)d_in[2];
    const float* enc_Wih  = (const float*)d_in[3];
    const float* enc_Whh  = (const float*)d_in[4];
    const float* enc_bih  = (const float*)d_in[5];
    const float* enc_bhh  = (const float*)d_in[6];
    const float* dec_emb  = (const float*)d_in[7];
    const float* attn_W   = (const float*)d_in[8];
    const float* attn_b   = (const float*)d_in[9];
    const float* v_W      = (const float*)d_in[10];
    const float* dec_Wih  = (const float*)d_in[11];
    const float* dec_Whh  = (const float*)d_in[12];
    const float* dec_bih  = (const float*)d_in[13];
    const float* dec_bhh  = (const float*)d_in[14];
    const float* fc_W     = (const float*)d_in[15];
    const float* fc_b     = (const float*)d_in[16];
    float*       out      = (float*)d_out;

    // ---- workspace carve-up (256B aligned) ----
    size_t off = 0;
    auto carve = [&](size_t bytes) -> void* {
        void* p = (char*)d_ws + off;
        off = (off + bytes + 255) & ~(size_t)255;
        return p;
    };
    bf16_t* encWihT  = (bf16_t*)carve((size_t)Ee  * G4H * 2);        // (512 ,4096)
    bf16_t* encWhhT  = (bf16_t*)carve((size_t)Hh  * G4H * 2);        // (1024,4096)
    bf16_t* decWihT  = (bf16_t*)carve((size_t)(Ee + Hh) * G4H * 2);  // (1536,4096)
    bf16_t* decWhhT  = (bf16_t*)carve((size_t)Hh  * G4H * 2);        // (1024,4096)
    bf16_t* attnWT   = (bf16_t*)carve((size_t)(2 * Hh) * Hh * 2);    // (2048,1024)
    bf16_t* fcWT     = (bf16_t*)carve((size_t)Hh  * VTt * 2);        // (1024,32000)
    bf16_t* embS     = (bf16_t*)carve((size_t)Bb * Ss * Ee * 2);     // bf16 acts
    bf16_t* embT     = (bf16_t*)carve((size_t)Bb * (Tt - 1) * Ee * 2);
    float*  enc_out  = (float*) carve((size_t)Bb * Ss * Hh * 4);
    bf16_t* encoutbf = (bf16_t*)carve((size_t)Bb * Ss * Hh * 2);
    float*  encproj  = (float*) carve((size_t)Bb * Ss * Hh * 4);
    float*  gatesXe  = (float*) carve((size_t)Bb * Ss * G4H * 4);    // 32 MB
    float*  gatesXd  = (float*) carve((size_t)Bb * (Tt - 1) * G4H * 4);
    bf16_t* hbf      = (bf16_t*)carve((size_t)Bb * Hh * 2);
    float*  cbuf     = (float*) carve((size_t)Bb * Hh * 4);
    float*  hproj    = (float*) carve((size_t)Bb * Hh * 4);
    bf16_t* ctxbf    = (bf16_t*)carve((size_t)Bb * Hh * 2);
    float*  gates    = (float*) carve((size_t)Bb * G4H * 4);
    float*  scores   = (float*) carve((size_t)Bb * Ss * 4);

    const dim3 tb(32, 8);
    auto tgrid = [](int N, int K) { return dim3((K + 31) / 32, (N + 31) / 32); };

    // ---- one-time weight transpose+convert to bf16, K-major ----
    transpose_cvt_bf16<<<tgrid(G4H, Ee),      tb, 0, stream>>>(enc_Wih, encWihT, G4H, Ee);
    transpose_cvt_bf16<<<tgrid(G4H, Hh),      tb, 0, stream>>>(enc_Whh, encWhhT, G4H, Hh);
    transpose_cvt_bf16<<<tgrid(G4H, Ee + Hh), tb, 0, stream>>>(dec_Wih, decWihT, G4H, Ee + Hh);
    transpose_cvt_bf16<<<tgrid(G4H, Hh),      tb, 0, stream>>>(dec_Whh, decWhhT, G4H, Hh);
    transpose_cvt_bf16<<<tgrid(Hh, 2 * Hh),   tb, 0, stream>>>(attn_W,  attnWT,  Hh, 2 * Hh);
    transpose_cvt_bf16<<<tgrid(VTt, Hh),      tb, 0, stream>>>(fc_W,    fcWT,    VTt, Hh);

    // ---- embedding gathers (bf16) ----
    {
        int totS = Bb * Ss * Ee;
        gather_emb_bf16<<<(totS + 255) / 256, 256, 0, stream>>>(src, Ss, Ss, enc_emb, embS, totS);
        int totT = Bb * (Tt - 1) * Ee;
        gather_emb_bf16<<<(totT + 255) / 256, 256, 0, stream>>>(tgt, Tt - 1, Tt, dec_emb, embT, totT);
    }

    // ---- init: h=c=0, out[:,0,:]=0 ----
    zero_f32<<<(Bb * Hh / 2 + 255) / 256, 256, 0, stream>>>((float*)hbf, Bb * Hh / 2);
    zero_f32<<<(Bb * Hh + 255) / 256, 256, 0, stream>>>(cbuf, Bb * Hh);
    zero_strided<<<(Bb * VTt + 255) / 256, 256, 0, stream>>>(out, Bb, VTt, (long long)Tt * VTt);

    auto gemm = [&](const bf16_t* A, int lda, const bf16_t* Bt, int ldb,
                    const float* bias, float* C, int ldc, int M, int N, int K, int acc) {
        dim3 grid(N / 256, M / 16);
        gemm_bf16_wmma<<<grid, 256, 0, stream>>>(A, lda, Bt, ldb, bias, C, ldc, K, acc);
    };

    // ---- batched input projections over all timesteps ----
    // gatesXe[b*S+t, :] = embS @ enc_Wih^T + bih      (M = 2048)
    gemm(embS, Ee, encWihT, G4H, enc_bih, gatesXe, G4H, Bb * Ss, G4H, Ee, 0);
    // gatesXd[b*(T-1)+t, :] = embT @ dec_Wih[:, :E]^T + bih   (M = 2016)
    gemm(embT, Ee, decWihT, G4H, dec_bih, gatesXd, G4H, Bb * (Tt - 1), G4H, Ee, 0);

    // ---- encoder: 64 sequential LSTM steps (recurrent GEMM only) ----
    for (int t = 0; t < Ss; ++t) {
        gemm(hbf, Hh, encWhhT, G4H, enc_bhh, gates, G4H, Bb, G4H, Hh, 0);
        lstm_cell<<<(Bb * Hh + 255) / 256, 256, 0, stream>>>(
            gates, gatesXe + (size_t)t * G4H, (long long)Ss * G4H,
            cbuf, hbf, enc_out + (size_t)t * Hh, encoutbf + (size_t)t * Hh, Ss * Hh);
    }

    // ---- encproj = enc_out @ Wa_e^T + attn_b  (computed once, M = 2048) ----
    gemm(encoutbf, Hh, attnWT + (size_t)Hh * Hh, Hh, attn_b,
         encproj, Hh, Bb * Ss, Hh, Hh, 0);

    // ---- decoder: 63 teacher-forced steps ----
    for (int t = 0; t < Tt - 1; ++t) {
        // hproj = h @ Wa_h^T
        gemm(hbf, Hh, attnWT, Hh, nullptr, hproj, Hh, Bb, Hh, Hh, 0);
        // scores + softmax + context (bf16)
        attn_scores<<<Bb * Ss, 256, 0, stream>>>(hproj, encproj, v_W, scores);
        softmax_context<<<Bb, 256, 0, stream>>>(scores, enc_out, ctxbf);
        // gates = ctx @ Wih[:, E:]^T ; += h @ Whh^T + bhh  (X part is batched)
        gemm(ctxbf, Hh, decWihT + (size_t)Ee * G4H, G4H, nullptr,
             gates, G4H, Bb, G4H, Hh, 0);
        gemm(hbf, Hh, decWhhT, G4H, dec_bhh, gates, G4H, Bb, G4H, Hh, 1);
        lstm_cell<<<(Bb * Hh + 255) / 256, 256, 0, stream>>>(
            gates, gatesXd + (size_t)t * G4H, (long long)(Tt - 1) * G4H,
            cbuf, hbf, nullptr, nullptr, 0);
        // pred -> out[:, t+1, :]
        gemm(hbf, Hh, fcWT, VTt, fc_b,
             out + (size_t)(t + 1) * VTt, Tt * VTt, Bb, VTt, Hh, 0);
    }
}
